// MultiHeadSelfAttention_13537736917691
// MI455X (gfx1250) — matrix operand order repaired and checked by reading, not compiled, not run on hardware
//
#include <hip/hip_runtime.h>
#include <math.h>

typedef __attribute__((ext_vector_type(16))) _Float16 v16h;
typedef __attribute__((ext_vector_type(8)))  float    v8f;
typedef __attribute__((ext_vector_type(4)))  unsigned int v4u;
typedef __attribute__((ext_vector_type(8)))  unsigned int v8u;

#define WMMA_F16(A, B, C) \
  __builtin_amdgcn_wmma_f32_16x16x32_f16(false, (A), false, (B), (short)0, (C), false, false)

static constexpr int B_   = 2;
static constexpr int N_   = 2048;
static constexpr int DIM_ = 1024;
static constexpr int H_   = 16;
static constexpr int HD_  = 64;       // head dim
static constexpr int ROWS = B_ * N_;  // 4096

// ---------------------------------------------------------------------------
// Tensor Data Mover: 2D tile load (global f16 -> LDS) with optional LDS row
// padding. D# built per CDNA5 ISA. Issued once per wave
// (EXEC is ignored by TENSOR_* ops); completion via TENSORcnt.
// ---------------------------------------------------------------------------
__device__ __forceinline__ void tdm_load_2d_f16(
    unsigned lds_byte_off, const void* gptr,
    unsigned tensor_d0, unsigned tensor_d1, unsigned long long stride0_elems,
    unsigned tile_d0, unsigned tile_d1,
    unsigned pad_interval_code, unsigned pad_amount_code) {
  unsigned long long ga = (unsigned long long)gptr;
  v4u g0;
  g0[0] = 1u;                                   // count=1, user descriptor
  g0[1] = lds_byte_off;                         // lds_addr
  g0[2] = (unsigned)ga;                         // global_addr[31:0]
  g0[3] = (unsigned)((ga >> 32) & 0x01FFFFFFu)  // global_addr[56:32]
          | (2u << 30);                         // type = 2 ("image")
  v8u g1;
  g1[0] = (1u << 16)                            // data_size = 1 (2 bytes)
          | (1u << 20)                          // pad_enable
          | (pad_interval_code << 22)
          | (pad_amount_code << 25);
  g1[1] = (tensor_d0 & 0xFFFFu) << 16;          // atomic_barrier=0 | tdim0 lo
  g1[2] = (tensor_d0 >> 16) | ((tensor_d1 & 0xFFFFu) << 16);
  g1[3] = (tensor_d1 >> 16) | (tile_d0 << 16);
  g1[4] = tile_d1;                              // tile_dim1 | tile_dim2(=0)<<16
  g1[5] = (unsigned)stride0_elems;              // tensor_dim0_stride[31:0]
  g1[6] = (unsigned)(stride0_elems >> 32) & 0xFFFFu;
  g1[7] = 0u;
  asm volatile("tensor_load_to_lds %0, %1" :: "s"(g0), "s"(g1) : "memory");
}

// ---------------------------------------------------------------------------
// Kernel 1: qkv = x @ w_qkv, fp32 inputs -> f16 output [4096, 3072]
// Tile: 128(M) x 64(N), K-step 32. 256 threads = 8 waves (4x2 wave grid).
// (fp32->f16 conversion happens during staging, so no TDM here.)
// ---------------------------------------------------------------------------
__global__ __launch_bounds__(256, 2)
void qkv_gemm_f16(const float* __restrict__ X, const float* __restrict__ W,
                  _Float16* __restrict__ Out) {
  constexpr int TM = 128, TN = 64, TK = 32, LDT = TK + 8;
  __shared__ _Float16 As[TM][LDT];    // [m][k]
  __shared__ _Float16 Bst[TN][LDT];   // transposed [n][k]

  const int tid  = threadIdx.x;
  const int lane = tid & 31;
  const int wave = tid >> 5;
  const int m0 = blockIdx.y * TM;
  const int n0 = blockIdx.x * TN;
  const int wm = (wave & 3) * 32;
  const int wn = (wave >> 2) * 32;

  v8f c00 = {}, c01 = {}, c10 = {}, c11 = {};

  const int arow = tid >> 1;
  const int acol = (tid & 1) * 16;
  const int brow = tid >> 3;
  const int bcol = (tid & 7) * 8;

  const int am = lane & 15;
  const int ak = (lane >> 4) * 8;
  const int bn = lane & 15;
  const int bk = (lane >> 4) * 8;

  for (int k0 = 0; k0 < DIM_; k0 += TK) {
    const float* ap = X + (size_t)(m0 + arow) * DIM_ + (k0 + acol);
#pragma unroll
    for (int i = 0; i < 4; ++i) {
      float4 v = ((const float4*)ap)[i];
      As[arow][acol + 4 * i + 0] = (_Float16)v.x;
      As[arow][acol + 4 * i + 1] = (_Float16)v.y;
      As[arow][acol + 4 * i + 2] = (_Float16)v.z;
      As[arow][acol + 4 * i + 3] = (_Float16)v.w;
    }
    const float* bp = W + (size_t)(k0 + brow) * (3 * DIM_) + (n0 + bcol);
#pragma unroll
    for (int i = 0; i < 2; ++i) {
      float4 v = ((const float4*)bp)[i];
      Bst[bcol + 4 * i + 0][brow] = (_Float16)v.x;
      Bst[bcol + 4 * i + 1][brow] = (_Float16)v.y;
      Bst[bcol + 4 * i + 2][brow] = (_Float16)v.z;
      Bst[bcol + 4 * i + 3][brow] = (_Float16)v.w;
    }
    if (k0 + TK < DIM_) {
      __builtin_prefetch(ap + TK, 0, 0);
      __builtin_prefetch(bp + (size_t)TK * (3 * DIM_), 0, 0);
    }
    __syncthreads();

    v16h a0, a1, b0, b1;
#pragma unroll
    for (int s = 0; s < 8; ++s) {
      a0[s]     = As[wm + am][ak + s];
      a0[s + 8] = As[wm + am][16 + ak + s];
      a1[s]     = As[wm + 16 + am][ak + s];
      a1[s + 8] = As[wm + 16 + am][16 + ak + s];
    }
#pragma unroll
    for (int s = 0; s < 16; ++s) {
      b0[s] = Bst[wn + bn][bk + s + (s >= 8 ? 8 : 0)];
      b1[s] = Bst[wn + 16 + bn][bk + s + (s >= 8 ? 8 : 0)];
    }
    c00 = WMMA_F16(a0, b0, c00);
    c01 = WMMA_F16(a0, b1, c01);
    c10 = WMMA_F16(a1, b0, c10);
    c11 = WMMA_F16(a1, b1, c11);
    __syncthreads();
  }

  const int cn = lane & 15;
  const int ro = (lane < 16) ? 0 : 8;
#pragma unroll
  for (int e = 0; e < 8; ++e) {
    size_t r0 = (size_t)(m0 + wm + ro + e) * (3 * DIM_);
    size_t r1 = (size_t)(m0 + wm + 16 + ro + e) * (3 * DIM_);
    Out[r0 + n0 + wn + cn]      = (_Float16)c00[e];
    Out[r0 + n0 + wn + 16 + cn] = (_Float16)c01[e];
    Out[r1 + n0 + wn + cn]      = (_Float16)c10[e];
    Out[r1 + n0 + wn + 16 + cn] = (_Float16)c11[e];
  }
}

// ---------------------------------------------------------------------------
// Kernel 2: flash attention. Block = 128 threads (4 waves), one (b,h) and a
// 64-query tile per block. K/V blocks staged to LDS by the Tensor Data Mover.
// ---------------------------------------------------------------------------
__global__ __launch_bounds__(128, 2)
void attn_flash_f16(const _Float16* __restrict__ QKV,  // [4096, 3072]
                    _Float16* __restrict__ AOut) {     // [4096, 1024]
  constexpr int LK = HD_ + 8;               // 72-half padded row
  __shared__ _Float16 Kld[64][LK];          // K block, [key][d]
  __shared__ _Float16 Vld[64][LK];          // V block, [key][d]
  __shared__ _Float16 Pld[4][16][LK];       // per-wave P strip

  const int tid  = threadIdx.x;
  const int lane = tid & 31;
  const int wave = tid >> 5;
  const int q0 = blockIdx.x * 64;
  const int bh = blockIdx.y;
  const int b  = bh >> 4;
  const int h  = bh & 15;

  const int am = lane & 15;
  const int ak = (lane >> 4) * 8;
  const int bn = lane & 15;
  const int bk = (lane >> 4) * 8;

  const size_t rowQ = (size_t)(b * N_ + q0 + wave * 16);

  // Q fragments held in registers for the whole key loop (A-layout).
  const _Float16* qrow = QKV + (rowQ + am) * (3 * DIM_) + (size_t)h * HD_;
  v16h aq0, aq1;
#pragma unroll
  for (int s = 0; s < 8; ++s) {
    aq0[s]     = qrow[ak + s];
    aq0[s + 8] = qrow[16 + ak + s];
    aq1[s]     = qrow[32 + ak + s];
    aq1[s + 8] = qrow[48 + ak + s];
  }

  v8f ot[4] = {};
  float mrow[8], lrow[8];
#pragma unroll
  for (int e = 0; e < 8; ++e) { mrow[e] = -INFINITY; lrow[e] = 0.0f; }

  const unsigned kld_off = (unsigned)(size_t)&Kld[0][0];
  const unsigned vld_off = (unsigned)(size_t)&Vld[0][0];

  for (int j = 0; j < N_ / 64; ++j) {
    const int k0 = j * 64;
    __syncthreads();  // all waves done reading previous K/V block
    if (wave == 0) {
      const _Float16* kg =
          QKV + (size_t)(b * N_ + k0) * (3 * DIM_) + DIM_ + h * HD_;
      // 64x64 f16 tile, row stride 3072 elems; LDS pad: every 32 dwords
      // (code 4) add 4 dwords (code 3) -> 72-half LDS rows.
      tdm_load_2d_f16(kld_off, kg, 64, 64, 3 * DIM_, 64, 64, 4, 3);
      tdm_load_2d_f16(vld_off, kg + DIM_, 64, 64, 3 * DIM_, 64, 64, 4, 3);
      __builtin_amdgcn_s_wait_tensorcnt(0);
    }
    __syncthreads();

    // S = Q @ K^T  (16 queries x 64 keys per wave)
    v8f st[4] = {};
#pragma unroll
    for (int ks = 0; ks < 2; ++ks) {
      v16h aq = ks ? aq1 : aq0;
#pragma unroll
      for (int t = 0; t < 4; ++t) {
        v16h bf;
#pragma unroll
        for (int s = 0; s < 16; ++s) bf[s] = Kld[t * 16 + bn][ks * 32 + bk + s + (s >= 8 ? 8 : 0)];
        st[t] = WMMA_F16(aq, bf, st[t]);
      }
    }

    // scale (1/sqrt(64)) + row max over the 64-key strip
    float rmax[8];
#pragma unroll
    for (int e = 0; e < 8; ++e) {
      float m = -INFINITY;
#pragma unroll
      for (int t = 0; t < 4; ++t) {
        st[t][e] *= 0.125f;
        m = fmaxf(m, st[t][e]);
      }
      rmax[e] = m;
    }
#pragma unroll
    for (int xm = 8; xm >= 1; xm >>= 1)
#pragma unroll
      for (int e = 0; e < 8; ++e)
        rmax[e] = fmaxf(rmax[e], __shfl_xor(rmax[e], xm, 32));

    // online softmax update
#pragma unroll
    for (int e = 0; e < 8; ++e) {
      float mnew = fmaxf(mrow[e], rmax[e]);
      float corr = __expf(mrow[e] - mnew);
      mrow[e] = mnew;
      lrow[e] *= corr;
#pragma unroll
      for (int t = 0; t < 4; ++t) ot[t][e] *= corr;
    }

    // P = exp(S - m): stage to per-wave LDS (C-layout -> A-layout reshape)
    float rsum[8];
#pragma unroll
    for (int e = 0; e < 8; ++e) rsum[e] = 0.0f;
    const int pro = (lane < 16) ? 0 : 8;
#pragma unroll
    for (int t = 0; t < 4; ++t)
#pragma unroll
      for (int e = 0; e < 8; ++e) {
        float p = __expf(st[t][e] - mrow[e]);
        rsum[e] += p;
        Pld[wave][pro + e][t * 16 + bn] = (_Float16)p;
      }
#pragma unroll
    for (int xm = 8; xm >= 1; xm >>= 1)
#pragma unroll
      for (int e = 0; e < 8; ++e) rsum[e] += __shfl_xor(rsum[e], xm, 32);
#pragma unroll
    for (int e = 0; e < 8; ++e) lrow[e] += rsum[e];

    // O += P @ V   (V stays [key][d]; B-frag slots stride over key rows)
#pragma unroll
    for (int ks = 0; ks < 2; ++ks) {
      v16h ap;
#pragma unroll
      for (int s = 0; s < 8; ++s) {
        ap[s]     = Pld[wave][am][ks * 32 + ak + s];
        ap[s + 8] = Pld[wave][am][ks * 32 + 16 + ak + s];
      }
#pragma unroll
      for (int t = 0; t < 4; ++t) {
        v16h bv;
#pragma unroll
        for (int s = 0; s < 16; ++s) bv[s] = Vld[ks * 32 + bk + s + (s >= 8 ? 8 : 0)][t * 16 + bn];
        ot[t] = WMMA_F16(ap, bv, ot[t]);
      }
    }
  }

  // normalize and store (f16) in [row, h*64 + d] layout for the out-proj GEMM
  const int ro = (lane < 16) ? 0 : 8;
#pragma unroll
  for (int t = 0; t < 4; ++t)
#pragma unroll
    for (int e = 0; e < 8; ++e) {
      float v = ot[t][e] / lrow[e];
      AOut[(rowQ + ro + e) * DIM_ + h * HD_ + t * 16 + bn] = (_Float16)v;
    }
}

// ---------------------------------------------------------------------------
// Kernel 3: out = attn @ w_out + b_out. f16 A staged by TDM, fp32 W converted
// during staging, fp32 out with fused bias.
// ---------------------------------------------------------------------------
__global__ __launch_bounds__(256, 2)
void out_gemm_f32(const _Float16* __restrict__ A, const float* __restrict__ W,
                  const float* __restrict__ bias, float* __restrict__ Out) {
  constexpr int TM = 128, TN = 64, TK = 32, LDT = TK + 8;
  __shared__ _Float16 As[TM][LDT];
  __shared__ _Float16 Bst[TN][LDT];

  const int tid  = threadIdx.x;
  const int lane = tid & 31;
  const int wave = tid >> 5;
  const int m0 = blockIdx.y * TM;
  const int n0 = blockIdx.x * TN;
  const int wm = (wave & 3) * 32;
  const int wn = (wave >> 2) * 32;

  v8f c00 = {}, c01 = {}, c10 = {}, c11 = {};

  const int brow = tid >> 3;
  const int bcol = (tid & 7) * 8;

  const int am = lane & 15;
  const int ak = (lane >> 4) * 8;
  const int bn = lane & 15;
  const int bk = (lane >> 4) * 8;

  const unsigned as_off = (unsigned)(size_t)&As[0][0];

  for (int k0 = 0; k0 < DIM_; k0 += TK) {
    if (wave == 0) {
      // 128-row x 32-half tile, row stride 1024 elems; LDS pad: every 16
      // dwords (code 3) add 4 dwords (code 3) -> 40-half LDS rows.
      tdm_load_2d_f16(as_off, A + (size_t)m0 * DIM_ + k0,
                      TK, TM, DIM_, TK, TM, 3, 3);
    }
    const float* bp = W + (size_t)(k0 + brow) * DIM_ + (n0 + bcol);
#pragma unroll
    for (int i = 0; i < 2; ++i) {
      float4 v = ((const float4*)bp)[i];
      Bst[bcol + 4 * i + 0][brow] = (_Float16)v.x;
      Bst[bcol + 4 * i + 1][brow] = (_Float16)v.y;
      Bst[bcol + 4 * i + 2][brow] = (_Float16)v.z;
      Bst[bcol + 4 * i + 3][brow] = (_Float16)v.w;
    }
    if (k0 + TK < DIM_) __builtin_prefetch(bp + (size_t)TK * DIM_, 0, 0);
    if (wave == 0) __builtin_amdgcn_s_wait_tensorcnt(0);
    __syncthreads();

    v16h a0, a1, b0, b1;
#pragma unroll
    for (int s = 0; s < 8; ++s) {
      a0[s]     = As[wm + am][ak + s];
      a0[s + 8] = As[wm + am][16 + ak + s];
      a1[s]     = As[wm + 16 + am][ak + s];
      a1[s + 8] = As[wm + 16 + am][16 + ak + s];
    }
#pragma unroll
    for (int s = 0; s < 16; ++s) {
      b0[s] = Bst[wn + bn][bk + s + (s >= 8 ? 8 : 0)];
      b1[s] = Bst[wn + 16 + bn][bk + s + (s >= 8 ? 8 : 0)];
    }
    c00 = WMMA_F16(a0, b0, c00);
    c01 = WMMA_F16(a0, b1, c01);
    c10 = WMMA_F16(a1, b0, c10);
    c11 = WMMA_F16(a1, b1, c11);
    __syncthreads();
  }

  const int cn = lane & 15;
  const int ro = (lane < 16) ? 0 : 8;
  const float bv0 = bias[n0 + wn + cn];
  const float bv1 = bias[n0 + wn + 16 + cn];
#pragma unroll
  for (int e = 0; e < 8; ++e) {
    size_t r0 = (size_t)(m0 + wm + ro + e) * DIM_;
    size_t r1 = (size_t)(m0 + wm + 16 + ro + e) * DIM_;
    Out[r0 + n0 + wn + cn]      = c00[e] + bv0;
    Out[r0 + n0 + wn + 16 + cn] = c01[e] + bv1;
    Out[r1 + n0 + wn + cn]      = c10[e] + bv0;
    Out[r1 + n0 + wn + 16 + cn] = c11[e] + bv1;
  }
}

extern "C" void kernel_launch(void* const* d_in, const int* in_sizes, int n_in,
                              void* d_out, int out_size, void* d_ws, size_t ws_size,
                              hipStream_t stream) {
  (void)in_sizes; (void)n_in; (void)out_size; (void)ws_size;
  const float* x     = (const float*)d_in[0];
  const float* w_qkv = (const float*)d_in[1];
  const float* w_out = (const float*)d_in[2];
  const float* b_out = (const float*)d_in[3];
  float* out = (float*)d_out;

  _Float16* qkv16  = (_Float16*)d_ws;                    // 4096*3072 f16 (24 MB)
  _Float16* attn16 = qkv16 + (size_t)ROWS * (3 * DIM_);  // 4096*1024 f16 (8 MB)

  dim3 g1(3 * DIM_ / 64, ROWS / 128);   // (48, 32)
  qkv_gemm_f16<<<g1, 256, 0, stream>>>(x, w_qkv, qkv16);

  dim3 g2(N_ / 64, B_ * H_);            // (32, 32)
  attn_flash_f16<<<g2, 128, 0, stream>>>(qkv16, attn16);

  dim3 g3(DIM_ / 64, ROWS / 128);       // (16, 32)
  out_gemm_f32<<<g3, 256, 0, stream>>>(attn16, w_out, b_out, out);
}
